// ImprovedContrastiveLoss_26431228739624
// MI455X (gfx1250) — matrix operand order repaired and checked by reading, not compiled, not run on hardware
//
#include <hip/hip_runtime.h>
#include <hip/hip_fp16.h>
#include <math.h>

// ImprovedContrastiveLoss, fused for MI455X (gfx1250, wave32, WMMA).
//
// Pipeline (all on `stream`):
//   1) zero sum buffers in ws
//   2) normalize rows of q,d and split each fp32 value into hi/lo fp16
//   3) fused GEMM(X @ Y^T) + exp(10*sim) + row/col-sum reduction kernels
//      using v_wmma_f32_16x16x32_f16 with 3-WMMA split-fp16 error correction.
//      qd: full matrix (row sums, col sums, diagonal).
//      qq/dd: symmetric -> strict upper triangle only; each e(i,j) feeds
//      rowsum[i] (row reduce) and rowsum[j] (col reduce, same array).
//   4) finalize: loss = -mean(log(pos) - log(partition))
//
// Workspace layout: q_hi,q_lo,d_hi,d_lo (4 x 16MB fp16) + 5*N floats of sums.

typedef __attribute__((ext_vector_type(16))) _Float16 v16h;
typedef __attribute__((ext_vector_type(8)))  _Float16 v8h;
typedef __attribute__((ext_vector_type(8)))  float    v8f;

#define NROWS 8192
#define DIM   1024
#define INVT  10.0f

// ---------------------------------------------------------------- zero sums
__global__ void zero_kernel(float* __restrict__ p, int n) {
  int i = blockIdx.x * blockDim.x + threadIdx.x;
  if (i < n) p[i] = 0.0f;
}

// ------------------------------------------- L2-normalize + split fp32->2xfp16
__global__ __launch_bounds__(256) void norm_split_kernel(
    const float* __restrict__ src, _Float16* __restrict__ hi,
    _Float16* __restrict__ lo) {
  __shared__ float red[8];
  const int row  = blockIdx.x;
  const int lane = threadIdx.x & 31;
  const int wave = threadIdx.x >> 5;
  const float* p = src + (size_t)row * DIM;

  float s = 0.0f;
  for (int c = threadIdx.x; c < DIM; c += 256) {
    float v = p[c];
    s += v * v;
  }
  for (int m = 16; m >= 1; m >>= 1) s += __shfl_xor(s, m, 32);
  if (lane == 0) red[wave] = s;
  __syncthreads();
  if (threadIdx.x == 0) {
    float t = 0.0f;
#pragma unroll
    for (int w = 0; w < 8; ++w) t += red[w];
    red[0] = t;
  }
  __syncthreads();
  const float rn = 1.0f / fmaxf(sqrtf(red[0]), 1e-12f);
  for (int c = threadIdx.x; c < DIM; c += 256) {
    float v = p[c] * rn;
    _Float16 h = (_Float16)v;
    hi[(size_t)row * DIM + c] = h;
    lo[(size_t)row * DIM + c] = (_Float16)(v - (float)h);
  }
}

// ------------------------------------------------------- WMMA helpers
__device__ __forceinline__ v16h cat8(v8h a, v8h b) {
  return __builtin_shufflevector(a, b, 0, 1, 2, 3, 4, 5, 6, 7, 8, 9, 10, 11,
                                 12, 13, 14, 15);
}

// 3-WMMA split-fp16 accumulate: c += Ahi*Bhi + Ahi*Blo + Alo*Bhi
__device__ __forceinline__ void mma3(v8f& c, const v16h& aH, const v16h& aL,
                                     const _Float16* __restrict__ brow_h,
                                     const _Float16* __restrict__ brow_l,
                                     int k, int half_) {
  v8h b0h = *(const v8h*)(brow_h + k + half_ * 8);
  v8h b1h = *(const v8h*)(brow_h + k + 16 + half_ * 8);
  v8h b0l = *(const v8h*)(brow_l + k + half_ * 8);
  v8h b1l = *(const v8h*)(brow_l + k + 16 + half_ * 8);
  v16h bH = cat8(b0h, b1h);
  v16h bL = cat8(b0l, b1l);
  c = __builtin_amdgcn_wmma_f32_16x16x32_f16(false, aH, false, bH, (short)0, c,
                                             false, false);
  c = __builtin_amdgcn_wmma_f32_16x16x32_f16(false, aH, false, bL, (short)0, c,
                                             false, false);
  c = __builtin_amdgcn_wmma_f32_16x16x32_f16(false, aL, false, bH, (short)0, c,
                                             false, false);
}

// Epilogue for one 16x16 tile: exp, diagonal/triangle fixup, row & col reduce.
__device__ __forceinline__ void post_tile(v8f& c, int rowBase, int cb, int l16,
                                          int half_, int sym,
                                          float* __restrict__ pos,
                                          float* __restrict__ scol16,
                                          int hasCol, v8f& rs) {
  float csloc = 0.0f;
  const int gn = cb + l16;
#pragma unroll
  for (int r = 0; r < 8; ++r) {
    float e = __expf(c[r] * INVT);
    const int gm = rowBase + r + half_ * 8;
    if (sym) {
      if (gn <= gm) e = 0.0f;          // strict upper triangle; diag excluded
    } else if (pos) {
      if (gm == gn) pos[gm] = e;       // positive pair exp_qd[i,i]
    }
    csloc += e;
    // row-sum across the 16 lanes of this half-wave (lanes share h)
    e += __shfl_xor(e, 1, 32);
    e += __shfl_xor(e, 2, 32);
    e += __shfl_xor(e, 4, 32);
    e += __shfl_xor(e, 8, 32);
    rs[r] += e;
  }
  if (hasCol) {
    csloc += __shfl_xor(csloc, 16, 32);  // combine M=0..7 (h=0) and M=8..15
    if (half_ == 0) atomicAdd(&scol16[l16], csloc);
  }
}

// ----------------- fused GEMM(X @ Y^T) + exp + row/col sums ------------------
// Block tile: 128 rows x 64 cols; wave w owns rows [blk.y*128+16w, +16),
// 4 accumulator tiles across 64 columns, K swept in steps of 32.
// sym==1: A==B symmetric; only strict upper triangle computed; colsum must
// alias rowsum so column reductions land in the same partition array.
__global__ __launch_bounds__(256) void gemm_exp_sum_kernel(
    const _Float16* __restrict__ Ah, const _Float16* __restrict__ Al,
    const _Float16* __restrict__ Bh, const _Float16* __restrict__ Bl,
    float* rowsum, float* colsum,
    float* __restrict__ pos, int sym) {
  __shared__ float scol[64];
  const int wave  = threadIdx.x >> 5;
  const int lane  = threadIdx.x & 31;
  const int half_ = lane >> 4;   // h: selects K sub-range per ISA layout
  const int l16   = lane & 15;

  const int rowBase0 = blockIdx.y * 128;
  const int rowBase  = rowBase0 + wave * 16;
  const int colBase  = blockIdx.x * 64;
  const int hasCol   = (colsum != nullptr);

  // Symmetric mode: drop blocks entirely on/below the diagonal (block-uniform,
  // before any barrier).
  if (sym && (colBase + 63 <= rowBase0)) return;

  if (threadIdx.x < 64) scol[threadIdx.x] = 0.0f;
  __syncthreads();

  // Symmetric mode: waves whose 16x64 strip has no strict-upper element skip
  // the K-loop (wave-uniform branch -> EXEC all-1s inside WMMA region).
  const bool waveActive = !sym || (colBase + 63 > rowBase);

  if (waveActive) {
    // A fragment rows (this wave's 16 rows; lane l16 = row m)
    const _Float16* arow_h = Ah + (size_t)(rowBase + l16) * DIM;
    const _Float16* arow_l = Al + (size_t)(rowBase + l16) * DIM;
    // B fragment rows for the 4 column tiles (lane l16 = column n)
    const _Float16* br_h0 = Bh + (size_t)(colBase + 0 * 16 + l16) * DIM;
    const _Float16* br_l0 = Bl + (size_t)(colBase + 0 * 16 + l16) * DIM;
    const _Float16* br_h1 = Bh + (size_t)(colBase + 1 * 16 + l16) * DIM;
    const _Float16* br_l1 = Bl + (size_t)(colBase + 1 * 16 + l16) * DIM;
    const _Float16* br_h2 = Bh + (size_t)(colBase + 2 * 16 + l16) * DIM;
    const _Float16* br_l2 = Bl + (size_t)(colBase + 2 * 16 + l16) * DIM;
    const _Float16* br_h3 = Bh + (size_t)(colBase + 3 * 16 + l16) * DIM;
    const _Float16* br_l3 = Bl + (size_t)(colBase + 3 * 16 + l16) * DIM;

    v8f c0 = {}, c1 = {}, c2 = {}, c3 = {};

    for (int k = 0; k < DIM; k += 32) {
      // A 16x32 f16 fragment: lane(m,h) holds K = h*8+{0..7}, 16+h*8+{0..7}
      v8h a0h = *(const v8h*)(arow_h + k + half_ * 8);
      v8h a1h = *(const v8h*)(arow_h + k + 16 + half_ * 8);
      v8h a0l = *(const v8h*)(arow_l + k + half_ * 8);
      v8h a1l = *(const v8h*)(arow_l + k + 16 + half_ * 8);
      v16h aH = cat8(a0h, a1h);
      v16h aL = cat8(a0l, a1l);

      mma3(c0, aH, aL, br_h0, br_l0, k, half_);
      mma3(c1, aH, aL, br_h1, br_l1, k, half_);
      mma3(c2, aH, aL, br_h2, br_l2, k, half_);
      mma3(c3, aH, aL, br_h3, br_l3, k, half_);
    }

    v8f rs = {};  // row-sum accumulators for this wave's 16 rows
    post_tile(c0, rowBase, colBase + 0 * 16, l16, half_, sym, pos,
              &scol[0 * 16], hasCol, rs);
    post_tile(c1, rowBase, colBase + 1 * 16, l16, half_, sym, pos,
              &scol[1 * 16], hasCol, rs);
    post_tile(c2, rowBase, colBase + 2 * 16, l16, half_, sym, pos,
              &scol[2 * 16], hasCol, rs);
    post_tile(c3, rowBase, colBase + 3 * 16, l16, half_, sym, pos,
              &scol[3 * 16], hasCol, rs);

    if (l16 == 0) {  // lanes 0 (rows +0..7) and 16 (rows +8..15) flush rows
#pragma unroll
      for (int r = 0; r < 8; ++r)
        atomicAdd(&rowsum[rowBase + r + half_ * 8], rs[r]);
    }
  }

  __syncthreads();
  if (hasCol && threadIdx.x < 64)
    atomicAdd(&colsum[colBase + threadIdx.x], scol[threadIdx.x]);
}

// ------------------------------------------------------------ final loss
__global__ __launch_bounds__(256) void finalize_kernel(
    const float* __restrict__ rs_qd, const float* __restrict__ cs_qd,
    const float* __restrict__ rs_qq, const float* __restrict__ cs_dd,
    const float* __restrict__ pos, float* __restrict__ out) {
  __shared__ float red[8];
  const int lane = threadIdx.x & 31;
  const int wave = threadIdx.x >> 5;
  float s = 0.0f;
  for (int i = threadIdx.x; i < NROWS; i += 256) {
    float part = rs_qd[i] + cs_qd[i] + rs_qq[i] + cs_dd[i];
    s += logf(pos[i]) - logf(part);
  }
  for (int m = 16; m >= 1; m >>= 1) s += __shfl_xor(s, m, 32);
  if (lane == 0) red[wave] = s;
  __syncthreads();
  if (threadIdx.x == 0) {
    float t = 0.0f;
#pragma unroll
    for (int w = 0; w < 8; ++w) t += red[w];
    out[0] = -t / (float)NROWS;
  }
}

// ---------------------------------------------------------------- launcher
extern "C" void kernel_launch(void* const* d_in, const int* in_sizes, int n_in,
                              void* d_out, int out_size, void* d_ws,
                              size_t ws_size, hipStream_t stream) {
  (void)in_sizes; (void)n_in; (void)out_size; (void)ws_size;
  const float* q = (const float*)d_in[0];
  const float* d = (const float*)d_in[1];

  char* ws = (char*)d_ws;
  const size_t matBytes = (size_t)NROWS * DIM * sizeof(_Float16);  // 16 MB
  _Float16* q_hi = (_Float16*)(ws + 0 * matBytes);
  _Float16* q_lo = (_Float16*)(ws + 1 * matBytes);
  _Float16* d_hi = (_Float16*)(ws + 2 * matBytes);
  _Float16* d_lo = (_Float16*)(ws + 3 * matBytes);
  float* sums  = (float*)(ws + 4 * matBytes);
  float* rs_qd = sums + 0 * NROWS;  // exp_qd row sums   (axis=1)
  float* cs_qd = sums + 1 * NROWS;  // exp_qd col sums   (axis=0)
  float* rs_qq = sums + 2 * NROWS;  // exp_qq row sums, diag excluded
  float* cs_dd = sums + 3 * NROWS;  // exp_dd col sums (= row sums by symmetry)
  float* pos   = sums + 4 * NROWS;  // exp_qd diagonal

  zero_kernel<<<(5 * NROWS + 255) / 256, 256, 0, stream>>>(sums, 5 * NROWS);
  norm_split_kernel<<<NROWS, 256, 0, stream>>>(q, q_hi, q_lo);
  norm_split_kernel<<<NROWS, 256, 0, stream>>>(d, d_hi, d_lo);

  dim3 grid(NROWS / 64, NROWS / 128);  // (128, 64)
  // q @ d^T : full matrix; row sums, col sums and the diagonal
  gemm_exp_sum_kernel<<<grid, 256, 0, stream>>>(q_hi, q_lo, d_hi, d_lo, rs_qd,
                                                cs_qd, pos, 0);
  // q @ q^T : symmetric; strict upper triangle, scattered to rows+cols
  gemm_exp_sum_kernel<<<grid, 256, 0, stream>>>(q_hi, q_lo, q_hi, q_lo, rs_qq,
                                                rs_qq, nullptr, 1);
  // d @ d^T : symmetric; col sums == row sums
  gemm_exp_sum_kernel<<<grid, 256, 0, stream>>>(d_hi, d_lo, d_hi, d_lo, cs_dd,
                                                cs_dd, nullptr, 1);

  finalize_kernel<<<1, 256, 0, stream>>>(rs_qd, cs_qd, rs_qq, cs_dd, pos,
                                         (float*)d_out);
}
